// TransformerBlock_62818191671515
// MI455X (gfx1250) — compile-verified, matrix operand order checked by reading, not run on hardware
//
#include <hip/hip_runtime.h>
#include <cstdint>

// ---------------------------------------------------------------- constants
constexpr int cS   = 1024;
constexpr int cDIM = 2048;
constexpr int cH   = 16;
constexpr int cKVH = 8;
constexpr int cHD  = 128;
constexpr int cFF  = 4096;
constexpr int cE   = 8;
constexpr float cEPS   = 1e-5f;
constexpr float cSCALE = 0.08838834764831845f; // 128^-0.5

// ---------------------------------------------------------------- WMMA types
typedef __bf16 bf16_t;
typedef bf16_t v16bf __attribute__((ext_vector_type(16)));
typedef float  v8f   __attribute__((ext_vector_type(8)));
typedef float  v4f   __attribute__((ext_vector_type(4)));

#define DEVFN static __device__ __forceinline__

// A/B fragment for 16x16x32 bf16 WMMA, source matrix fp32.
// Layout (ISA 7.12.2): lane L<16 -> row L, K = 0..7 (v0-3) and 16..23 (v4-7);
// lane L>=16 -> row L-16, K = 8..15 and 24..31.  Caller pre-offsets the
// pointer by row*ld + (lane>>4)*8, so here we just read [0..7] and [16..23].
// Native cast -> v_cvt_pk_bf16_f32 (RNE).
DEVFN v16bf load_frag_nt(const float* __restrict__ p) {
  v4f x0 = *(const v4f*)(p);
  v4f x1 = *(const v4f*)(p + 4);
  v4f x2 = *(const v4f*)(p + 16);
  v4f x3 = *(const v4f*)(p + 20);
  v16bf f;
#pragma unroll
  for (int j = 0; j < 4; ++j) {
    f[j]      = (bf16_t)x0[j];
    f[4 + j]  = (bf16_t)x1[j];
    f[8 + j]  = (bf16_t)x2[j];
    f[12 + j] = (bf16_t)x3[j];
  }
  return f;
}

// B fragment when B is (K x N) row-major: lane gives column n, K strided by ldb.
DEVFN v16bf load_frag_nn(const float* __restrict__ p, long ldb) {
  v16bf f;
#pragma unroll
  for (int j = 0; j < 8; ++j) f[j] = (bf16_t)p[(long)j * ldb];
#pragma unroll
  for (int j = 0; j < 8; ++j) f[8 + j] = (bf16_t)p[(long)(16 + j) * ldb];
  return f;
}

DEVFN v8f wmma_bf16(v16bf a, v16bf b, v8f c) {
  return __builtin_amdgcn_wmma_f32_16x16x32_bf16(false, a, false, b, (short)0, c,
                                                 false, false);
}

// Store one 16x16 f32 D tile: lane holds rows r+8*(lane>>4), col = lane&15.
DEVFN void store_sub(float* __restrict__ C, long ldc,
                     const float* __restrict__ R, long ldr,
                     int row0, int col0, v8f acc, float alpha, int lane) {
  const int col = col0 + (lane & 15);
  const int rb  = row0 + ((lane >> 4) << 3);
#pragma unroll
  for (int r = 0; r < 8; ++r) {
    float v = alpha * acc[r];
    if (R) v += R[(long)(rb + r) * ldr + col];
    C[(long)(rb + r) * ldc + col] = v;
  }
}

// ---------------------------------------------------------------- generic GEMM
// C[z] = alpha * A[z] @ op(B[z/bdiv]) + R[z]   (op = B^T if TRANSB)
// Block: 256 threads = 8 waves; 4 waves in M x 2 in N; wave tile 32x32.
// Block tile 128(M) x 64(N). All dims must be multiples of the tile (they are).
template <bool TRANSB>
__global__ __launch_bounds__(256)
void gemm_kernel(const float* __restrict__ A, long lda, long Abs,
                 const float* __restrict__ B, long ldb, long Bbs, int bdiv,
                 float* __restrict__ C, long ldc, long Cbs,
                 const float* __restrict__ R, long ldr, long Rbs,
                 int K, float alpha) {
  const int z = blockIdx.z;
  A += (long)z * Abs;
  B += (long)(z / bdiv) * Bbs;
  C += (long)z * Cbs;
  if (R) R += (long)z * Rbs;

  const int wave = threadIdx.x >> 5;
  const int lane = threadIdx.x & 31;
  const int wm = wave & 3;
  const int wn = wave >> 2;
  const int tileM = blockIdx.y * 128 + wm * 32;
  const int tileN = blockIdx.x * 64 + wn * 32;
  const int lr = lane & 15;
  const int lh = lane >> 4;

  const float* a0 = A + (long)(tileM + lr) * lda + lh * 8;
  const float* a1 = A + (long)(tileM + 16 + lr) * lda + lh * 8;
  const float *b0, *b1;
  if (TRANSB) {
    b0 = B + (long)(tileN + lr) * ldb + lh * 8;
    b1 = B + (long)(tileN + 16 + lr) * ldb + lh * 8;
  } else {
    b0 = B + (long)(lh * 8) * ldb + (tileN + lr);
    b1 = B + (long)(lh * 8) * ldb + (tileN + 16 + lr);
  }

  v8f c00 = {}, c01 = {}, c10 = {}, c11 = {};
  for (int k0 = 0; k0 < K; k0 += 32) {
    v16bf fa0 = load_frag_nt(a0 + k0);
    v16bf fa1 = load_frag_nt(a1 + k0);
    v16bf fb0, fb1;
    if (TRANSB) {
      fb0 = load_frag_nt(b0 + k0);
      fb1 = load_frag_nt(b1 + k0);
      if (k0 + 32 < K) {  // stream weights one K-step ahead
        __builtin_prefetch(b0 + k0 + 32, 0, 1);
        __builtin_prefetch(b1 + k0 + 32, 0, 1);
      }
    } else {
      fb0 = load_frag_nn(b0 + (long)k0 * ldb, ldb);
      fb1 = load_frag_nn(b1 + (long)k0 * ldb, ldb);
    }
    c00 = wmma_bf16(fa0, fb0, c00);
    c01 = wmma_bf16(fa0, fb1, c01);
    c10 = wmma_bf16(fa1, fb0, c10);
    c11 = wmma_bf16(fa1, fb1, c11);
  }
  store_sub(C, ldc, R, ldr, tileM,      tileN,      c00, alpha, lane);
  store_sub(C, ldc, R, ldr, tileM,      tileN + 16, c01, alpha, lane);
  store_sub(C, ldc, R, ldr, tileM + 16, tileN,      c10, alpha, lane);
  store_sub(C, ldc, R, ldr, tileM + 16, tileN + 16, c11, alpha, lane);
}

// ---------------------------------------------------------------- RMSNorm
__global__ __launch_bounds__(256)
void rmsnorm_kernel(const float* __restrict__ x, const float* __restrict__ w,
                    float* __restrict__ out) {
  const int t = blockIdx.x;
  const int tid = threadIdx.x;
  __shared__ float red[256];
  const float* xr = x + (long)t * cDIM;
  float s = 0.f;
#pragma unroll
  for (int j = 0; j < cDIM / 256; ++j) {
    float v = xr[tid + j * 256];
    s += v * v;
  }
  red[tid] = s;
  __syncthreads();
  for (int o = 128; o > 0; o >>= 1) {
    if (tid < o) red[tid] += red[tid + o];
    __syncthreads();
  }
  const float inv = rsqrtf(red[0] / (float)cDIM + cEPS);
#pragma unroll
  for (int j = 0; j < cDIM / 256; ++j) {
    int i = tid + j * 256;
    out[(long)t * cDIM + i] = xr[i] * inv * w[i];
  }
}

// ---------------------------------------------------------------- RoPE (in place)
__global__ __launch_bounds__(256)
void rope_kernel(float* __restrict__ t, const float* __restrict__ cb,
                 const float* __restrict__ sb, int heads, int total) {
  int gid = blockIdx.x * 256 + threadIdx.x;
  if (gid >= total) return;
  const int pair = gid & 63;
  const int sh = gid >> 6;
  const int h = sh % heads;
  const int s = sh / heads;
  const float c = cb[s * 64 + pair];
  const float sn = sb[s * 64 + pair];
  float* p = t + ((long)s * heads + h) * cHD + pair * 2;
  const float t0 = p[0], t1 = p[1];
  p[0] = t0 * c - t1 * sn;
  p[1] = t0 * sn + t1 * c;
}

// ---------------------------------------------------------------- row softmax (1024 wide, in place)
__global__ __launch_bounds__(256)
void softmax_kernel(float* __restrict__ sc) {
  float* row = sc + (long)blockIdx.x * 1024;
  const int tid = threadIdx.x;
  __shared__ float red[256];
  float v0 = row[tid], v1 = row[tid + 256], v2 = row[tid + 512], v3 = row[tid + 768];
  float mx = fmaxf(fmaxf(v0, v1), fmaxf(v2, v3));
  red[tid] = mx;
  __syncthreads();
  for (int o = 128; o > 0; o >>= 1) {
    if (tid < o) red[tid] = fmaxf(red[tid], red[tid + o]);
    __syncthreads();
  }
  mx = red[0];
  __syncthreads();
  v0 = __expf(v0 - mx); v1 = __expf(v1 - mx); v2 = __expf(v2 - mx); v3 = __expf(v3 - mx);
  red[tid] = v0 + v1 + v2 + v3;
  __syncthreads();
  for (int o = 128; o > 0; o >>= 1) {
    if (tid < o) red[tid] += red[tid + o];
    __syncthreads();
  }
  const float inv = __builtin_amdgcn_rcpf(red[0]);
  row[tid] = v0 * inv; row[tid + 256] = v1 * inv;
  row[tid + 512] = v2 * inv; row[tid + 768] = v3 * inv;
}

// ---------------------------------------------------------------- gating
__global__ void zero_counts_kernel(int* counts) {
  if (threadIdx.x < cE) counts[threadIdx.x] = 0;
}

__global__ __launch_bounds__(256)
void gate_kernel(const float* __restrict__ hn, const float* __restrict__ gw,
                 float* __restrict__ wout, int* __restrict__ counts,
                 int* __restrict__ lists) {
  const int t = blockIdx.x;
  const int tid = threadIdx.x;
  __shared__ float red[256];
  __shared__ float logits[cE];
  const float* xr = hn + (long)t * cDIM;
  for (int e = 0; e < cE; ++e) {
    float s = 0.f;
    for (int i = tid; i < cDIM; i += 256) s += xr[i] * gw[(long)e * cDIM + i];
    red[tid] = s;
    __syncthreads();
    for (int o = 128; o > 0; o >>= 1) {
      if (tid < o) red[tid] += red[tid + o];
      __syncthreads();
    }
    if (tid == 0) logits[e] = red[0];
    __syncthreads();
  }
  if (tid == 0) {
    float mx = logits[0];
    for (int e = 1; e < cE; ++e) mx = fmaxf(mx, logits[e]);
    float p[cE];
    for (int e = 0; e < cE; ++e) p[e] = __expf(logits[e] - mx);
    int i0 = 0;
    for (int e = 1; e < cE; ++e) if (p[e] > p[i0]) i0 = e;
    int i1 = (i0 == 0) ? 1 : 0;
    for (int e = 0; e < cE; ++e) if (e != i0 && p[e] > p[i1]) i1 = e;
    const float ws = p[i0] + p[i1];
    wout[t * 2 + 0] = p[i0] / ws;
    wout[t * 2 + 1] = p[i1] / ws;
    int pos0 = atomicAdd(&counts[i0], 1);
    lists[i0 * cS + pos0] = t * 2;
    int pos1 = atomicAdd(&counts[i1], 1);
    lists[i1 * cS + pos1] = t * 2 + 1;
  }
}

// ---------------------------------------------------------------- MoE GEMM 1
// up[slot, ff] = silu(hn[t] @ W1[e]^T) * (hn[t] @ W3[e]^T), rows gathered via lists.
// 8 waves: 4 in M (32 rows each -> 128 tokens) x 2 in N (16 ff cols each).
// Large M-tile = fewer passes over the streamed expert weights (~2 avg).
__global__ __launch_bounds__(256)
void moe_ffn1_kernel(const float* __restrict__ hn, const float* __restrict__ W1,
                     const float* __restrict__ W3, const int* __restrict__ counts,
                     const int* __restrict__ lists, float* __restrict__ up) {
  const int e = blockIdx.z;
  const int count = counts[e];
  if ((int)(blockIdx.y * 128) >= count) return;
  const int wave = threadIdx.x >> 5;
  const int lane = threadIdx.x & 31;
  const int wm = wave & 3, wn = wave >> 2;
  const int lr = lane & 15, lh = lane >> 4;
  const int m0 = blockIdx.y * 128 + wm * 32;
  const int ff0 = blockIdx.x * 32 + wn * 16;
  const int* lst = lists + e * cS;

  int q0 = m0 + lr;        if (q0 > count - 1) q0 = count - 1;
  int q1 = m0 + 16 + lr;   if (q1 > count - 1) q1 = count - 1;
  const float* a0 = hn + (long)(lst[q0] >> 1) * cDIM + lh * 8;
  const float* a1 = hn + (long)(lst[q1] >> 1) * cDIM + lh * 8;
  const float* b1 = W1 + ((long)e * cFF + ff0 + lr) * cDIM + lh * 8;
  const float* b3 = W3 + ((long)e * cFF + ff0 + lr) * cDIM + lh * 8;

  v8f c1a = {}, c1b = {}, c3a = {}, c3b = {};
  for (int k0 = 0; k0 < cDIM; k0 += 32) {
    v16bf fa0 = load_frag_nt(a0 + k0);
    v16bf fa1 = load_frag_nt(a1 + k0);
    v16bf fb1 = load_frag_nt(b1 + k0);
    v16bf fb3 = load_frag_nt(b3 + k0);
    if (k0 + 32 < cDIM) {  // stream expert weights ahead
      __builtin_prefetch(b1 + k0 + 32, 0, 1);
      __builtin_prefetch(b3 + k0 + 32, 0, 1);
    }
    c1a = wmma_bf16(fa0, fb1, c1a);
    c1b = wmma_bf16(fa1, fb1, c1b);
    c3a = wmma_bf16(fa0, fb3, c3a);
    c3b = wmma_bf16(fa1, fb3, c3b);
  }
  const int col = ff0 + lr;
#pragma unroll
  for (int sub = 0; sub < 2; ++sub) {
    v8f u1 = sub ? c1b : c1a;
    v8f u3 = sub ? c3b : c3a;
#pragma unroll
    for (int r = 0; r < 8; ++r) {
      const int pos = m0 + sub * 16 + r + 8 * lh;
      if (pos < count) {
        const int slot = lst[pos];
        const float x1 = u1[r];
        // silu via v_rcp_f32 (bf16-accuracy workload; avoid precise-div sequence)
        const float sig = __builtin_amdgcn_rcpf(1.0f + __expf(-x1));
        up[(long)slot * cFF + col] = x1 * sig * u3[r];
      }
    }
  }
}

// ---------------------------------------------------------------- MoE GEMM 2
// y[slot, d] = up[slot] @ W2[e]^T.
// 8 waves: 4 in M (128 tokens) x 2 in N (32 cols each). Block tile 128x64.
__global__ __launch_bounds__(256)
void moe_ffn2_kernel(const float* __restrict__ up, const float* __restrict__ W2,
                     const int* __restrict__ counts, const int* __restrict__ lists,
                     float* __restrict__ y) {
  const int e = blockIdx.z;
  const int count = counts[e];
  if ((int)(blockIdx.y * 128) >= count) return;
  const int wave = threadIdx.x >> 5;
  const int lane = threadIdx.x & 31;
  const int wm = wave & 3, wn = wave >> 2;
  const int lr = lane & 15, lh = lane >> 4;
  const int m0 = blockIdx.y * 128 + wm * 32;
  const int n0 = blockIdx.x * 64 + wn * 32;
  const int* lst = lists + e * cS;

  int q0 = m0 + lr;        if (q0 > count - 1) q0 = count - 1;
  int q1 = m0 + 16 + lr;   if (q1 > count - 1) q1 = count - 1;
  const float* a0 = up + (long)lst[q0] * cFF + lh * 8;
  const float* a1 = up + (long)lst[q1] * cFF + lh * 8;
  const float* b0 = W2 + ((long)e * cDIM + n0 + lr) * cFF + lh * 8;
  const float* b1 = W2 + ((long)e * cDIM + n0 + 16 + lr) * cFF + lh * 8;

  v8f c00 = {}, c01 = {}, c10 = {}, c11 = {};
  for (int k0 = 0; k0 < cFF; k0 += 32) {
    v16bf fa0 = load_frag_nt(a0 + k0);
    v16bf fa1 = load_frag_nt(a1 + k0);
    v16bf fb0 = load_frag_nt(b0 + k0);
    v16bf fb1 = load_frag_nt(b1 + k0);
    if (k0 + 32 < cFF) {  // stream expert weights ahead
      __builtin_prefetch(b0 + k0 + 32, 0, 1);
      __builtin_prefetch(b1 + k0 + 32, 0, 1);
    }
    c00 = wmma_bf16(fa0, fb0, c00);
    c01 = wmma_bf16(fa0, fb1, c01);
    c10 = wmma_bf16(fa1, fb0, c10);
    c11 = wmma_bf16(fa1, fb1, c11);
  }
#pragma unroll
  for (int sub = 0; sub < 2; ++sub) {
#pragma unroll
    for (int nsub = 0; nsub < 2; ++nsub) {
      v8f acc = sub ? (nsub ? c11 : c10) : (nsub ? c01 : c00);
      const int col = n0 + nsub * 16 + lr;
#pragma unroll
      for (int r = 0; r < 8; ++r) {
        const int pos = m0 + sub * 16 + r + 8 * lh;
        if (pos < count) {
          y[(long)lst[pos] * cDIM + col] = acc[r];
        }
      }
    }
  }
}

// ---------------------------------------------------------------- combine
__global__ __launch_bounds__(256)
void combine_kernel(const float* __restrict__ h, const float* __restrict__ y,
                    const float* __restrict__ w, float* __restrict__ out) {
  const long i = (long)blockIdx.x * 256 + threadIdx.x;
  const int t = (int)(i / cDIM);
  const int d = (int)(i % cDIM);
  out[i] = h[i] + w[t * 2 + 0] * y[(long)(t * 2 + 0) * cDIM + d]
               + w[t * 2 + 1] * y[(long)(t * 2 + 1) * cDIM + d];
}

// ---------------------------------------------------------------- launch
extern "C" void kernel_launch(void* const* d_in, const int* in_sizes, int n_in,
                              void* d_out, int out_size, void* d_ws, size_t ws_size,
                              hipStream_t stream) {
  (void)in_sizes; (void)n_in; (void)out_size; (void)ws_size;
  const float* x    = (const float*)d_in[0];
  const float* fcos = (const float*)d_in[1];
  const float* fsin = (const float*)d_in[2];
  const float* wq   = (const float*)d_in[3];
  const float* wk   = (const float*)d_in[4];
  const float* wv   = (const float*)d_in[5];
  const float* wo   = (const float*)d_in[6];
  const float* anw  = (const float*)d_in[7];
  const float* fnw  = (const float*)d_in[8];
  const float* gw   = (const float*)d_in[9];
  const float* W1   = (const float*)d_in[10];
  const float* W2   = (const float*)d_in[11];
  const float* W3   = (const float*)d_in[12];
  float* out = (float*)d_out;

  // workspace carve (floats)
  float* f = (float*)d_ws;
  float* xn = f;                      f += (long)cS * cDIM;       // 2M
  float* qb = f;                      f += (long)cS * cH * cHD;   // 2M
  float* kb = f;                      f += (long)cS * cKVH * cHD; // 1M
  float* vb = f;                      f += (long)cS * cKVH * cHD; // 1M
  float* sc = f;                      f += (long)cH * cS * cS;    // 16M
  float* ao = f;                      f += (long)cS * cH * cHD;   // 2M
  float* hb = f;                      f += (long)cS * cDIM;       // 2M
  float* hnb = f;                     f += (long)cS * cDIM;       // 2M
  float* up = f;                      f += (long)cS * 2 * cFF;    // 8M
  float* yb = f;                      f += (long)cS * 2 * cDIM;   // 4M
  float* wgt = f;                     f += (long)cS * 2;
  int* counts = (int*)f;
  int* lists  = counts + cE;

  // 1) attn rmsnorm
  rmsnorm_kernel<<<cS, 256, 0, stream>>>(x, anw, xn);

  // 2) q/k/v = xn @ w{q,k,v}^T
  gemm_kernel<true><<<dim3(cH * cHD / 64, cS / 128, 1), 256, 0, stream>>>(
      xn, cDIM, 0, wq, cDIM, 0, 1, qb, cH * cHD, 0, nullptr, 0, 0, cDIM, 1.0f);
  gemm_kernel<true><<<dim3(cKVH * cHD / 64, cS / 128, 1), 256, 0, stream>>>(
      xn, cDIM, 0, wk, cDIM, 0, 1, kb, cKVH * cHD, 0, nullptr, 0, 0, cDIM, 1.0f);
  gemm_kernel<true><<<dim3(cKVH * cHD / 64, cS / 128, 1), 256, 0, stream>>>(
      xn, cDIM, 0, wv, cDIM, 0, 1, vb, cKVH * cHD, 0, nullptr, 0, 0, cDIM, 1.0f);

  // 3) RoPE
  rope_kernel<<<(cS * cH * 64 + 255) / 256, 256, 0, stream>>>(qb, fcos, fsin, cH,
                                                              cS * cH * 64);
  rope_kernel<<<(cS * cKVH * 64 + 255) / 256, 256, 0, stream>>>(kb, fcos, fsin, cKVH,
                                                                cS * cKVH * 64);

  // 4) scores[h] = SCALE * q_h @ k_{h/2}^T   (batch z = head, GQA bdiv=2)
  gemm_kernel<true><<<dim3(cS / 64, cS / 128, cH), 256, 0, stream>>>(
      qb, cH * cHD, cHD, kb, cKVH * cHD, cHD, 2, sc, cS, (long)cS * cS,
      nullptr, 0, 0, cHD, cSCALE);

  // 5) softmax over keys
  softmax_kernel<<<cH * cS, 256, 0, stream>>>(sc);

  // 6) attn out[h] = probs_h @ v_{h/2}   (NN: B is K x N with row stride KVH*HD)
  gemm_kernel<false><<<dim3(cHD / 64, cS / 128, cH), 256, 0, stream>>>(
      sc, cS, (long)cS * cS, vb, cKVH * cHD, cHD, 2, ao, cH * cHD, cHD,
      nullptr, 0, 0, cS, 1.0f);

  // 7) h = x + ao @ wo^T
  gemm_kernel<true><<<dim3(cDIM / 64, cS / 128, 1), 256, 0, stream>>>(
      ao, cH * cHD, 0, wo, cH * cHD, 0, 1, hb, cDIM, 0, x, cDIM, 0,
      cH * cHD, 1.0f);

  // 8) ffn rmsnorm
  rmsnorm_kernel<<<cS, 256, 0, stream>>>(hb, fnw, hnb);

  // 9) gating + expert token lists
  zero_counts_kernel<<<1, 32, 0, stream>>>(counts);
  gate_kernel<<<cS, 256, 0, stream>>>(hnb, gw, wgt, counts, lists);

  // 10) MoE FFN (128-token tiles -> ~2 streaming passes over expert weights)
  moe_ffn1_kernel<<<dim3(cFF / 32, cS / 128, cE), 256, 0, stream>>>(
      hnb, W1, W3, counts, lists, up);
  moe_ffn2_kernel<<<dim3(cDIM / 64, cS / 128, cE), 256, 0, stream>>>(
      up, W2, counts, lists, yb);

  // 11) out = h + sum_k w_k * y_k
  combine_kernel<<<(cS * cDIM) / 256, 256, 0, stream>>>(hb, yb, wgt, out);
}
